// RecurrentLayer_22823456211636
// MI455X (gfx1250) — matrix operand order repaired and checked by reading, not compiled, not run on hardware
//
#include <hip/hip_runtime.h>
#include <hip/hip_bf16.h>

// ---------------------------------------------------------------------------
// Leaky RNN, persistent cooperative kernel for MI455X (gfx1250, wave32, WMMA)
//
//   v_t  = (1-a) v_{t-1} + a * ( [fr_{t-1} | u_t] @ [W_hid^T ; W_in^T] + b )
//   fr_t = relu(v_t)
//
// 16 workgroups x 256 threads (8 waves). WG w owns output columns
// [w*32, w*32+32). Its 640x32 bf16 weight slice is pre-swizzled into the
// WMMA B-operand lane layout and held in LDS for all 1024 steps. Each wave
// owns one 16x16 tile; its f32 state `v` stays in VGPRs for the whole
// sequence. fr is exchanged through a double-buffered bf16 L2-resident
// buffer with one device-wide barrier per step.
// ---------------------------------------------------------------------------

typedef __attribute__((ext_vector_type(16))) __bf16 v16bf;
typedef __attribute__((ext_vector_type(8)))  __bf16 v8bf;
typedef __attribute__((ext_vector_type(8)))  float  v8f;
typedef __attribute__((ext_vector_type(4)))  float  v4f;

#define SEQ    1024
#define BATCH  64
#define INDIM  128
#define HID    512
#define NWG    16                 // workgroups (== max cluster size)
#define NCOL   (HID / NWG)        // 32 output columns per WG
#define KTOT   (HID + INDIM)      // 640 combined K
#define NKC    (KTOT / 32)        // 20 K-chunks of 32
#define NKH    (HID / 32)         // 16 K-chunks from fr
#define ALPHA  0.1f

// workspace layout
#define WS_CNT_BYTES 4096                          // SEQ ints (per-step barrier)
#define FRBUF_ELEMS  (BATCH * HID)                 // one fr buffer (bf16)

// ---------------------------------------------------------------------------
// init: zero per-step barrier counters and fr_{-1} (buffer 0)
// ---------------------------------------------------------------------------
__global__ void rnn_init_kernel(int* __restrict__ counters,
                                __bf16* __restrict__ frbuf0) {
  int i = blockIdx.x * blockDim.x + threadIdx.x;
  if (i < SEQ) counters[i] = 0;
  for (int j = i; j < FRBUF_ELEMS; j += gridDim.x * blockDim.x)
    frbuf0[j] = (__bf16)0.0f;
}

// ---------------------------------------------------------------------------
// persistent recurrence kernel
// ---------------------------------------------------------------------------
__launch_bounds__(256)
__global__ void rnn_persistent_kernel(const float* __restrict__ input,   // (SEQ,BATCH,INDIM)
                                      const float* __restrict__ W_in,    // (HID,INDIM)
                                      const float* __restrict__ b_in,    // (HID)
                                      const float* __restrict__ W_hid,   // (HID,HID)
                                      const float* __restrict__ b_hid,   // (HID)
                                      float*       __restrict__ out,     // (SEQ,BATCH,HID)
                                      __bf16*      __restrict__ frbuf,   // 2 * BATCH*HID bf16
                                      int*         __restrict__ counters)
{
  // B-operand slice, pre-swizzled: [NKC][2 n-groups][32 lanes][16 halves]
  __shared__ __bf16 Wlds[NKC * 2 * 32 * 16];     // 40 KB

  const int wg   = blockIdx.x;          // 0..15
  const int tid  = threadIdx.x;
  const int lane = tid & 31;
  const int wave = tid >> 5;            // 0..7
  const int mi   = wave & 3;            // M tile (batch rows  mi*16..)
  const int ni   = wave >> 2;           // N tile within WG (0..1)

  // ---- one-time: load weight slice, swizzled into WMMA B lane layout ----
  // B layout (16-bit, 32x16): halves 0-7 = K koff..koff+7, 8-15 = K+16..,
  // koff = 0 for lanes 0-15, 8 for lanes 16-31 (matches A).
  for (int idx = tid; idx < NKC * 2 * 32 * 16; idx += 256) {
    int h  = idx & 15;
    int ln = (idx >> 4) & 31;
    int g  = (idx >> 9) & 1;
    int c  = idx >> 10;
    int k  = c * 32 + (h & 7) + ((ln & 16) ? 8 : 0) + ((h & 8) ? 16 : 0);
    int n  = wg * NCOL + g * 16 + (ln & 15);
    float w = (k < HID) ? W_hid[(size_t)n * HID + k]
                        : W_in[(size_t)n * INDIM + (k - HID)];
    Wlds[idx] = (__bf16)w;
  }
  __syncthreads();

  // ---- per-lane constants ----
  const int n_glob = wg * NCOL + ni * 16 + (lane & 15);    // C/D column
  const float bias = b_hid[n_glob] + b_in[n_glob];
  const int m_row  = mi * 16 + (lane & 15);                // A row for loads
  const int khalf  = (lane & 16) ? 8 : 0;                  // A half-K offset
  const int m_off  = (lane & 16) ? 8 : 0;                  // C/D row offset

  v8f v = {};                                              // persistent state

  for (int t = 0; t < SEQ; ++t) {
    const __bf16* frprev = frbuf + (size_t)(t & 1) * FRBUF_ELEMS;
    __bf16*       frnext = frbuf + (size_t)((t + 1) & 1) * FRBUF_ELEMS;

    v8f acc = {};

    // ---- recurrent part: K chunks 0..15 over fr_{t-1} (bf16, L2-resident) --
    // A layout (16-bit, 16x32): halves 0-7 = K koff..koff+7, 8-15 = K+16..
    const __bf16* arow = frprev + (size_t)m_row * HID;
    #pragma unroll
    for (int c = 0; c < NKH; ++c) {
      v8bf alo = *(const v8bf*)(arow + c * 32 + khalf);
      v8bf ahi = *(const v8bf*)(arow + c * 32 + khalf + 16);
      v16bf a  = __builtin_shufflevector(alo, ahi,
                   0,1,2,3,4,5,6,7,8,9,10,11,12,13,14,15);
      const __bf16* bp = Wlds + ((c * 2 + ni) * 32 + lane) * 16;
      v8bf blo = *(const v8bf*)(bp);
      v8bf bhi = *(const v8bf*)(bp + 8);
      v16bf b  = __builtin_shufflevector(blo, bhi,
                   0,1,2,3,4,5,6,7,8,9,10,11,12,13,14,15);
      acc = __builtin_amdgcn_wmma_f32_16x16x32_bf16(
                false, a, false, b, (short)0, acc, false, false);
    }

    // ---- input projection part: K chunks 16..19 over u_t (fp32 -> bf16) ---
    const float* urow = input + ((size_t)t * BATCH + m_row) * INDIM;
    #pragma unroll
    for (int c = 0; c < NKC - NKH; ++c) {
      int i0 = c * 32 + khalf;
      v4f f0 = *(const v4f*)(urow + i0);
      v4f f1 = *(const v4f*)(urow + i0 + 4);
      v4f f2 = *(const v4f*)(urow + i0 + 16);
      v4f f3 = *(const v4f*)(urow + i0 + 20);
      v16bf a;
      #pragma unroll
      for (int e = 0; e < 4; ++e) {
        a[e]      = (__bf16)f0[e];
        a[4 + e]  = (__bf16)f1[e];
        a[8 + e]  = (__bf16)f2[e];
        a[12 + e] = (__bf16)f3[e];
      }
      const __bf16* bp = Wlds + (((NKH + c) * 2 + ni) * 32 + lane) * 16;
      v8bf blo = *(const v8bf*)(bp);
      v8bf bhi = *(const v8bf*)(bp + 8);
      v16bf b  = __builtin_shufflevector(blo, bhi,
                   0,1,2,3,4,5,6,7,8,9,10,11,12,13,14,15);
      acc = __builtin_amdgcn_wmma_f32_16x16x32_bf16(
                false, a, false, b, (short)0, acc, false, false);
    }

    // ---- leaky integration + ReLU; state stays in VGPRs -------------------
    // C/D layout: VGPR r -> M = mi*16 + r (+8 for lanes 16-31), N = n_glob.
    #pragma unroll
    for (int r = 0; r < 8; ++r) {
      float vn = (1.0f - ALPHA) * v[r] + ALPHA * (acc[r] + bias);
      v[r] = vn;
      float fr = vn > 0.0f ? vn : 0.0f;
      int m = mi * 16 + r + m_off;
      out[((size_t)t * BATCH + m) * HID + n_glob] = fr;        // fp32 states
      frnext[(size_t)m * HID + n_glob] = (__bf16)fr;           // bf16 exchange
    }

    // ---- device-wide step barrier ----------------------------------------
    __threadfence();            // release fr stores to device scope
    __syncthreads();
    if (tid == 0) {
      atomicAdd(&counters[t], 1);
      volatile int* vc = counters;
      while (vc[t] < NWG) __builtin_amdgcn_s_sleep(1);
    }
    // Fast path if the runtime dispatches the 16 WGs as one cluster
    // (s_barrier_signal -3 / s_barrier_wait -3); NOP otherwise.
    __builtin_amdgcn_s_cluster_barrier();
    __syncthreads();
    __threadfence();            // acquire: drop stale cached fr lines
  }
}

// ---------------------------------------------------------------------------
extern "C" void kernel_launch(void* const* d_in, const int* in_sizes, int n_in,
                              void* d_out, int out_size, void* d_ws, size_t ws_size,
                              hipStream_t stream) {
  const float* input = (const float*)d_in[0];
  const float* W_in  = (const float*)d_in[1];
  const float* b_in  = (const float*)d_in[2];
  const float* W_hid = (const float*)d_in[3];
  const float* b_hid = (const float*)d_in[4];
  float* out = (float*)d_out;

  int*    counters = (int*)d_ws;
  __bf16* frbuf    = (__bf16*)((char*)d_ws + WS_CNT_BYTES);

  rnn_init_kernel<<<64, 256, 0, stream>>>(counters, frbuf);
  rnn_persistent_kernel<<<NWG, 256, 0, stream>>>(
      input, W_in, b_in, W_hid, b_hid, out, frbuf, counters);
}
